// StructLNN_53068615909518
// MI455X (gfx1250) — matrix operand order repaired and checked by reading, not compiled, not run on hardware
//
#include <hip/hip_runtime.h>
#include <hip/hip_bf16.h>
#include <stdint.h>

// MI455X / gfx1250, wave32. WMMA bf16 16x16x32, f32 accumulate.
// Workspace layout (bytes), total ~807 MB:
//   [0,          512K)   Wx  in B-fragment layout (rows 0..255 of [Wff1;Wff2;Wta;Wtb] cols cat)
//   [512K,       1M  )   Wh  in B-fragment layout (rows 256..511)
//   [1M,         129M)   feat bf16 [B*S, 256] row-major
//   [129M,       641M)   pre  bf16, C-fragment stream layout [mb][t][nt][lane][8]
//   [641M,       769M)   seq  bf16 [mb][t][m][256]

typedef __attribute__((ext_vector_type(16))) __bf16 v16bf;
typedef __attribute__((ext_vector_type(8)))  float  v8f;

struct F32x8pack { uint4 a, b; };   // 32 bytes == one v16bf fragment

__device__ __forceinline__ unsigned short f2bf(float f) {
    union { float f; unsigned u; } v; v.f = f;
    unsigned r = v.u + 0x7FFFu + ((v.u >> 16) & 1u);   // RNE
    return (unsigned short)(r >> 16);
}
__device__ __forceinline__ float bf2f(unsigned h) {
    union { unsigned u; float f; } v; v.u = h << 16; return v.f;
}
__device__ __forceinline__ float wred(float v) {
    #pragma unroll
    for (int o = 16; o; o >>= 1) v += __shfl_xor(v, o, 32);
    return v;
}
__device__ __forceinline__ float fsig(float x) {
    return __builtin_amdgcn_rcpf(1.0f + __expf(-x));
}
// Raw LDS byte offset of a generic pointer into __shared__ (for async-LDS VDST operand)
__device__ __forceinline__ unsigned ldsOffsetOf(const void* p) {
    return (unsigned)(unsigned long long)(const __attribute__((address_space(3))) void*)p;
}
// CDNA5 async global->LDS copy, 16B per lane, tracked by ASYNCcnt.
__device__ __forceinline__ void asyncLdsLoadB128(unsigned ldsOff, const void* gptr) {
    asm volatile("global_load_async_to_lds_b128 %0, %1, off"
                 :: "v"(ldsOff), "v"(gptr) : "memory");
}
__device__ __forceinline__ void waitAsync4() {
    asm volatile("s_wait_asynccnt 0x4" ::: "memory");
}
__device__ __forceinline__ void waitAsync0() {
    asm volatile("s_wait_asynccnt 0x0" ::: "memory");
}

// ---------------------------------------------------------------------------
// K0: pack W_ff1/W_ff2/W_ta/W_tb (f32 [512,256]) into bf16 B-fragment layout.
// B-frag (32x16 bf16): lane l (r=l&15, hl=l>>4), elem e: K = kt*32 + hl*8 + (e&7) + (e>>3)*16,
// N = nt*16 + r.  Storage: [kt][nt][lane][e].
// ---------------------------------------------------------------------------
__global__ __launch_bounds__(256) void k0_pack_w(
    const float* __restrict__ wff1, const float* __restrict__ wff2,
    const float* __restrict__ wta,  const float* __restrict__ wtb,
    unsigned short* __restrict__ wx, unsigned short* __restrict__ wh) {
    unsigned idx   = blockIdx.x * 256u + threadIdx.x;      // 0..524287
    unsigned which = idx >> 18;                            // 0=wx (x half), 1=wh (h half)
    unsigned i     = idx & 0x3FFFFu;
    unsigned e  = i & 15u, lane = (i >> 4) & 31u, nt = (i >> 9) & 63u, kt = (i >> 15) & 7u;
    unsigned r  = lane & 15u, hl = lane >> 4;
    unsigned K  = kt * 32u + hl * 8u + (e & 7u) + ((e >> 3) << 4) + which * 256u;
    unsigned n  = nt * 16u + r, g = n >> 8, c = n & 255u;
    const float* W = (g == 0) ? wff1 : (g == 1) ? wff2 : (g == 2) ? wta : wtb;
    (which ? wh : wx)[i] = f2bf(W[K * 256u + c]);
}

// ---------------------------------------------------------------------------
// K1: feat = SiLU(LN(LN(x) @ proj_W + proj_b)), bf16 out. One wave per row.
// ---------------------------------------------------------------------------
__global__ __launch_bounds__(256) void k1_ln_proj(
    const float* __restrict__ x, const float* __restrict__ g0, const float* __restrict__ b0,
    const float* __restrict__ pW, const float* __restrict__ pb,
    const float* __restrict__ g1, const float* __restrict__ b1,
    unsigned short* __restrict__ feat) {
    __shared__ float xs[8][68];
    int wave = threadIdx.x >> 5, lane = threadIdx.x & 31;
    unsigned row = blockIdx.x * 8u + (unsigned)wave;        // 262144 rows
    const float* xr = x + (size_t)row * 67u;
    float v0 = xr[lane];
    float v1 = xr[lane + 32];                               // 32..63 < 67 always valid
    int  c2  = (lane + 64 < 67);
    float v2 = c2 ? xr[lane + 64] : 0.0f;
    float m  = wred(v0 + v1 + v2) * (1.0f / 67.0f);
    float d0 = v0 - m, d1 = v1 - m, d2 = c2 ? (v2 - m) : 0.0f;
    float var = wred(d0 * d0 + d1 * d1 + d2 * d2) * (1.0f / 67.0f);
    float rs  = rsqrtf(var + 1e-5f);
    xs[wave][lane]      = d0 * rs * g0[lane]      + b0[lane];
    xs[wave][lane + 32] = d1 * rs * g0[lane + 32] + b0[lane + 32];
    if (c2) xs[wave][lane + 64] = d2 * rs * g0[lane + 64] + b0[lane + 64];
    __syncthreads();

    int n0 = lane * 8;
    float acc[8];
    #pragma unroll
    for (int j = 0; j < 8; j++) acc[j] = pb[n0 + j];
    for (int d = 0; d < 67; d++) {
        float xd = xs[wave][d];
        const float* wr = pW + d * 256 + n0;
        #pragma unroll
        for (int j = 0; j < 8; j++) acc[j] = fmaf(xd, wr[j], acc[j]);
    }
    float s2 = 0.f, q2 = 0.f;
    #pragma unroll
    for (int j = 0; j < 8; j++) { s2 += acc[j]; q2 += acc[j] * acc[j]; }
    s2 = wred(s2); q2 = wred(q2);
    float m2  = s2 * (1.0f / 256.0f);
    float rs2 = rsqrtf(q2 * (1.0f / 256.0f) - m2 * m2 + 1e-5f);
    unsigned out[4];
    #pragma unroll
    for (int j = 0; j < 4; j++) {
        float ya = (acc[2*j]   - m2) * rs2 * g1[n0 + 2*j]   + b1[n0 + 2*j];
        float yb = (acc[2*j+1] - m2) * rs2 * g1[n0 + 2*j+1] + b1[n0 + 2*j+1];
        ya *= fsig(ya); yb *= fsig(yb);
        out[j] = (unsigned)f2bf(ya) | ((unsigned)f2bf(yb) << 16);
    }
    *reinterpret_cast<uint4*>(feat + (size_t)row * 256u + n0) =
        make_uint4(out[0], out[1], out[2], out[3]);
}

// ---------------------------------------------------------------------------
// K2: pre = feat @ Wx + gate-bias   ([B*S,256] x [256,1024], bf16 WMMA).
// One wave per (mb,t,ngroup-of-4-Ntiles). Output stored as bf16 C-fragment
// stream [mb][t][nt][lane][8] so the scan kernel loads one b128 per tile.
// ---------------------------------------------------------------------------
__global__ __launch_bounds__(256) void k2_xgemm(
    const unsigned short* __restrict__ feat, const unsigned short* __restrict__ wx,
    const float* __restrict__ bff1, const float* __restrict__ bff2,
    const float* __restrict__ bta,  const float* __restrict__ btb,
    unsigned short* __restrict__ pre) {
    int wave = threadIdx.x >> 5, lane = threadIdx.x & 31;
    unsigned task = blockIdx.x * 8u + (unsigned)wave;       // 262144 tasks
    unsigned ng = task & 15u, tmp = task >> 4;
    unsigned t  = tmp & 2047u, mb = tmp >> 11;
    unsigned r  = lane & 15u, hl = lane >> 4;
    const unsigned short* arow = feat + (size_t)((mb * 16u + r) * 2048u + t) * 256u;

    v8f acc[4];
    #pragma unroll
    for (int j = 0; j < 4; j++) {
        unsigned n = (ng * 4u + j) * 16u + r, g = n >> 8, c = n & 255u;
        float bv = (g == 0 ? bff1 : g == 1 ? bff2 : g == 2 ? bta : btb)[c];
        #pragma unroll
        for (int i = 0; i < 8; i++) acc[j][i] = bv;
    }
    for (int kt = 0; kt < 8; kt++) {
        F32x8pack af;
        af.a = *reinterpret_cast<const uint4*>(arow + kt * 32 + hl * 8);
        af.b = *reinterpret_cast<const uint4*>(arow + kt * 32 + hl * 8 + 16);
        v16bf A = __builtin_bit_cast(v16bf, af);
        #pragma unroll
        for (int j = 0; j < 4; j++) {
            const unsigned short* bp = wx + (((unsigned)kt * 64u + ng * 4u + j) * 32u + lane) * 16u;
            F32x8pack bf;
            bf.a = *reinterpret_cast<const uint4*>(bp);
            bf.b = *reinterpret_cast<const uint4*>(bp + 8);
            v16bf Bm = __builtin_bit_cast(v16bf, bf);
            acc[j] = __builtin_amdgcn_wmma_f32_16x16x32_bf16(
                false, A, false, Bm, (short)0, acc[j], false, false);
        }
    }
    #pragma unroll
    for (int j = 0; j < 4; j++) {
        unsigned nt = ng * 4u + j, pd[4];
        #pragma unroll
        for (int q = 0; q < 4; q++)
            pd[q] = (unsigned)f2bf(acc[j][2*q]) | ((unsigned)f2bf(acc[j][2*q+1]) << 16);
        *reinterpret_cast<uint4*>(pre + (size_t)(((mb * 2048u + t) * 64u + nt) * 32u + lane) * 8u) =
            make_uint4(pd[0], pd[1], pd[2], pd[3]);
    }
}

// ---------------------------------------------------------------------------
// K3: CfC scan. 8 workgroups (one 16-batch-row tile each), 16 waves, 80KB
// dynamic LDS. Wave w owns h-cols [16w,16w+16): N-tiles {w,16+w,32+w,48+w}
// => all 4 gates for the same columns land in the same lanes => in-register
// gate combine. Wh B-fragments resident in VGPRs for all 2048 steps; h
// double-buffered in LDS (A-frag layout, one barrier/step); pre bias
// fragments double-buffered in LDS via GLOBAL_LOAD_ASYNC_TO_LDS_B128 with
// s_wait_asynccnt so HBM latency is off the sequential critical path.
// ---------------------------------------------------------------------------
__global__ __launch_bounds__(512) void k3_scan(
    const unsigned short* __restrict__ wh, const unsigned short* __restrict__ pre,
    const float* __restrict__ dt, unsigned short* __restrict__ seq) {
    extern __shared__ __align__(16) char smem3[];
    unsigned short* hbuf = (unsigned short*)smem3;                 // 2 x 4096 ushort (16KB)
    unsigned short* preS = (unsigned short*)(smem3 + 16384);       // 2 x 16 x 4 x 256 ushort (64KB)
    int wave = threadIdx.x >> 5, lane = threadIdx.x & 31;
    int mb   = blockIdx.x;
    unsigned r = lane & 15u, hl = lane >> 4;

    // Resident Wh fragments: 4 gates x 8 k-tiles = 256 VGPRs
    v16bf whB[4][8];
    #pragma unroll
    for (int j = 0; j < 4; j++) {
        #pragma unroll
        for (int kt = 0; kt < 8; kt++) {
            const unsigned short* bp =
                wh + (((unsigned)kt * 64u + (unsigned)(j * 16 + wave)) * 32u + lane) * 16u;
            F32x8pack bw;
            bw.a = *reinterpret_cast<const uint4*>(bp);
            bw.b = *reinterpret_cast<const uint4*>(bp + 8);
            whB[j][kt] = __builtin_bit_cast(v16bf, bw);
        }
    }
    // h0 = 0
    *reinterpret_cast<uint4*>(&hbuf[threadIdx.x * 8]) = make_uint4(0, 0, 0, 0);

    const unsigned short* preW = pre + (size_t)mb * 2048u * 64u * 32u * 8u;
    // Per-wave LDS byte offsets for the 4 pre fragments (buffer 0); buffer 1 is +32KB.
    unsigned preOff0 = ldsOffsetOf(preS) + ((unsigned)wave * 4u) * 512u + (unsigned)lane * 16u;

    // Kick off async copies of step-0 bias fragments into buffer 0.
    {
        const unsigned short* pt = preW;
        #pragma unroll
        for (int j = 0; j < 4; j++)
            asyncLdsLoadB128(preOff0 + (unsigned)j * 512u,
                             pt + (((unsigned)(j * 16 + wave)) * 32u + lane) * 8u);
    }
    __syncthreads();

    // Scatter constants for writing h_new (D-layout reg -> A-frag LDS), col c=16w+r
    unsigned c    = 16u * (unsigned)wave + r;
    unsigned kt2  = (unsigned)wave >> 1;
    unsigned e2   = (r & 7u) | (((unsigned)wave & 1u) << 3);
    unsigned hl2  = (r >> 3) & 1u;

    for (int t = 0; t < 2048; t++) {
        // Fire async copies for step t+1 into the other LDS buffer, then wait
        // for the older (in-order) batch of 4 to land.
        if (t + 1 < 2048) {
            const unsigned short* ptn = preW + (size_t)(t + 1) * (64u * 32u * 8u);
            unsigned dstOff = preOff0 + (((unsigned)(t + 1) & 1u) << 15);   // +32KB if odd buf
            #pragma unroll
            for (int j = 0; j < 4; j++)
                asyncLdsLoadB128(dstOff + (unsigned)j * 512u,
                                 ptn + (((unsigned)(j * 16 + wave)) * 32u + lane) * 8u);
            waitAsync4();
        } else {
            waitAsync0();
        }

        // Accumulator init = bias fragments from LDS (bf16 -> f32)
        const unsigned short* myPre =
            preS + (((((unsigned)t & 1u) * 16u + (unsigned)wave) * 4u) * 256u) + (unsigned)lane * 8u;
        v8f acc[4];
        #pragma unroll
        for (int j = 0; j < 4; j++) {
            uint4 pv = *reinterpret_cast<const uint4*>(myPre + j * 256);
            acc[j][0] = bf2f(pv.x & 0xFFFFu); acc[j][1] = bf2f(pv.x >> 16);
            acc[j][2] = bf2f(pv.y & 0xFFFFu); acc[j][3] = bf2f(pv.y >> 16);
            acc[j][4] = bf2f(pv.z & 0xFFFFu); acc[j][5] = bf2f(pv.z >> 16);
            acc[j][6] = bf2f(pv.w & 0xFFFFu); acc[j][7] = bf2f(pv.w >> 16);
        }
        float dtv = dt[(mb * 16 + (lane & 15)) * 2048 + t] * 10.0f;   // TIME_SCALE

        const unsigned short* hb = hbuf + (t & 1) * 4096;
        #pragma unroll
        for (int kt = 0; kt < 8; kt++) {
            F32x8pack ha;
            ha.a = *reinterpret_cast<const uint4*>(&hb[kt * 512 + lane * 16]);
            ha.b = *reinterpret_cast<const uint4*>(&hb[kt * 512 + lane * 16 + 8]);
            v16bf A = __builtin_bit_cast(v16bf, ha);
            #pragma unroll
            for (int j = 0; j < 4; j++)
                acc[j] = __builtin_amdgcn_wmma_f32_16x16x32_bf16(
                    false, A, false, whB[j][kt], (short)0, acc[j], false, false);
        }
        unsigned short* hn = hbuf + ((t + 1) & 1) * 4096;
        unsigned short* sq = seq + (size_t)((mb * 2048u + (unsigned)t) * 16u) * 256u;
        #pragma unroll
        for (int i = 0; i < 8; i++) {
            unsigned m = (unsigned)i + 8u * hl;                 // batch row in tile
            float dtt = __shfl(dtv, (int)m, 32);
            float ff1 = tanhf(acc[0][i]);
            float ff2 = tanhf(acc[1][i]);
            float ti  = fsig(acc[2][i] * dtt + acc[3][i]);
            float h   = ff1 + ti * (ff2 - ff1);
            unsigned short h16 = f2bf(h);
            hn[kt2 * 512u + (m + 16u * hl2) * 16u + e2] = h16;  // next-step A-frag
            sq[m * 256u + c] = h16;                             // for head kernel
        }
        __syncthreads();   // h writes visible; all LDS reads of old buffers done
    }
}

// ---------------------------------------------------------------------------
// K4: logits = LN(seq) @ head_W + head_bias. One wave per (b,t) row.
// ---------------------------------------------------------------------------
__global__ __launch_bounds__(256) void k4_head(
    const unsigned short* __restrict__ seq, const float* __restrict__ hg,
    const float* __restrict__ hb, const float* __restrict__ hw,
    const float* __restrict__ hbias, float* __restrict__ out) {
    int wave = threadIdx.x >> 5, lane = threadIdx.x & 31;
    unsigned q = blockIdx.x * 8u + (unsigned)wave;            // 0..262143 (mb,t,m packed)
    uint4 pv = *reinterpret_cast<const uint4*>(seq + (size_t)q * 256u + lane * 8);
    float xv[8] = { bf2f(pv.x & 0xFFFFu), bf2f(pv.x >> 16),
                    bf2f(pv.y & 0xFFFFu), bf2f(pv.y >> 16),
                    bf2f(pv.z & 0xFFFFu), bf2f(pv.z >> 16),
                    bf2f(pv.w & 0xFFFFu), bf2f(pv.w >> 16) };
    float s = 0.f, ss = 0.f;
    #pragma unroll
    for (int j = 0; j < 8; j++) { s += xv[j]; ss += xv[j] * xv[j]; }
    s = wred(s); ss = wred(ss);
    float m  = s * (1.0f / 256.0f);
    float rs = rsqrtf(ss * (1.0f / 256.0f) - m * m + 1e-5f);
    int n0 = lane * 8;
    float y = 0.f;
    #pragma unroll
    for (int j = 0; j < 8; j++)
        y += ((xv[j] - m) * rs * hg[n0 + j] + hb[n0 + j]) * hw[n0 + j];
    y = wred(y);
    if (lane == 0) {
        unsigned mb = q >> 15, mm = q & 15u, t = (q >> 4) & 2047u;
        out[(size_t)(mb * 16u + mm) * 2048u + t] = y + hbias[0];
    }
}

// ---------------------------------------------------------------------------
extern "C" void kernel_launch(void* const* d_in, const int* in_sizes, int n_in,
                              void* d_out, int out_size, void* d_ws, size_t ws_size,
                              hipStream_t stream) {
    (void)in_sizes; (void)n_in; (void)out_size; (void)ws_size;
    const float* x     = (const float*)d_in[0];
    const float* dt    = (const float*)d_in[1];
    const float* lng   = (const float*)d_in[2];
    const float* lnb   = (const float*)d_in[3];
    const float* projW = (const float*)d_in[4];
    const float* projb = (const float*)d_in[5];
    const float* lnpg  = (const float*)d_in[6];
    const float* lnpb  = (const float*)d_in[7];
    const float* wff1  = (const float*)d_in[8];
    const float* bff1  = (const float*)d_in[9];
    const float* wff2  = (const float*)d_in[10];
    const float* bff2  = (const float*)d_in[11];
    const float* wta   = (const float*)d_in[12];
    const float* bta   = (const float*)d_in[13];
    const float* wtb   = (const float*)d_in[14];
    const float* btb   = (const float*)d_in[15];
    const float* hg    = (const float*)d_in[16];
    const float* hbv   = (const float*)d_in[17];
    const float* hw    = (const float*)d_in[18];
    const float* hbias = (const float*)d_in[19];
    float* out = (float*)d_out;

    char* w = (char*)d_ws;
    unsigned short* wx   = (unsigned short*)(w);                              // 512 KB
    unsigned short* wh   = wx + 262144;                                       // 512 KB
    unsigned short* feat = (unsigned short*)(w + (1u << 20));                 // 128 MB
    unsigned short* pre  = (unsigned short*)(w + (1u << 20) + 134217728ull);  // 512 MB
    unsigned short* seq  = (unsigned short*)(w + (1u << 20) + 134217728ull + 536870912ull); // 128 MB

    k0_pack_w<<<2048, 256, 0, stream>>>(wff1, wff2, wta, wtb, wx, wh);
    k1_ln_proj<<<32768, 256, 0, stream>>>(x, lng, lnb, projW, projb, lnpg, lnpb, feat);
    k2_xgemm<<<32768, 256, 0, stream>>>(feat, wx, bff1, bff2, bta, btb, pre);
    k3_scan<<<8, 512, 81920, stream>>>(wh, pre, dt, seq);
    k4_head<<<32768, 256, 0, stream>>>(seq, hg, hbv, hw, hbias, out);
}
